// GATv2Conv_3401614098970
// MI455X (gfx1250) — matrix operand order repaired and checked by reading, not compiled, not run on hardware
//
#include <hip/hip_runtime.h>
#include <cstdint>
#include <cstddef>

typedef __bf16 v16bf __attribute__((ext_vector_type(16)));
typedef __bf16 v8bf  __attribute__((ext_vector_type(8)));
typedef float  v8f   __attribute__((ext_vector_type(8)));
typedef float  v4f   __attribute__((ext_vector_type(4)));

#define D_IN      256
#define HD        256              // H * D_OUT
#define NCOL      512              // [src-proj | dst-proj]
#define NHEAD     4
#define NEG_SLOPE 0.2f

// ---------------------------------------------------------------------------
// 1a) Convert concatenated weights W_src||W_dst (f32, [512,256]) to bf16.
// ---------------------------------------------------------------------------
__global__ void wconv_kernel(const float* __restrict__ Wsrc,
                             const float* __restrict__ Wdst,
                             __bf16* __restrict__ Wb) {
    int i = blockIdx.x * blockDim.x + threadIdx.x;        // < 512*256
    const int half = HD * D_IN;                            // 65536
    float v = (i < half) ? Wsrc[i] : Wdst[i - half];
    Wb[i] = (__bf16)v;
}

// ---------------------------------------------------------------------------
// 1b) Convert feat (f32 [N,256]) to bf16 copy, 4 elems per thread.
// ---------------------------------------------------------------------------
__global__ void fconv_kernel(const float* __restrict__ feat,
                             __bf16* __restrict__ featb, int nElem4) {
    int i = blockIdx.x * blockDim.x + threadIdx.x;        // quads
    if (i >= nElem4) return;
    v4f f = *(const v4f*)(feat + (size_t)i * 4);
    __bf16 b[4] = {(__bf16)f.x, (__bf16)f.y, (__bf16)f.z, (__bf16)f.w};
    *(uint2*)(featb + (size_t)i * 4) = *(const uint2*)b;
}

// ---------------------------------------------------------------------------
// 2) Init: out = 0, denom = 0, maxbuf = ordered-int(-inf) = INT_MIN.
// ---------------------------------------------------------------------------
__global__ void init_kernel(int* __restrict__ maxbuf, float* __restrict__ denom,
                            float* __restrict__ out, int nNodes) {
    int i = blockIdx.x * blockDim.x + threadIdx.x;
    if (i < nNodes * HD)    out[i] = 0.0f;
    if (i < nNodes * NHEAD) { maxbuf[i] = (int)0x80000000; denom[i] = 0.0f; }
}

// ---------------------------------------------------------------------------
// 3) Projection GEMM via WMMA bf16.
//    Block = 8 waves = one 16-row strip (mtile = blockIdx.x).
//    Each wave computes a 16x64 strip: 4 column tiles share one A fragment,
//    4 WMMAs per K-step, 32 WMMAs per wave.
//    wave32 layouts used here (A/B k order is contiguous, not the ISA table order):
//      A (16x32 bf16): lanes 0-15 -> M=lane, K=0..15 contiguous;
//                      lanes 16-31 -> M=lane-16, K=16..31
//      B (32x16 bf16): lanes 0-15 -> N=lane, K=0..15 contiguous;
//                      lanes 16-31 -> N=lane-16, K=16..31
//      C (16x16 f32):  VGPR r -> M = r + 8*(lane>>4), N = lane&15
// ---------------------------------------------------------------------------
__global__ void __launch_bounds__(256)
proj_wmma_kernel(const __bf16* __restrict__ featb,
                 const __bf16* __restrict__ Wb,
                 const float* __restrict__ bsrc,
                 const float* __restrict__ bdst,
                 float* __restrict__ P, int nNodes) {
    const int lane  = threadIdx.x & 31;
    const int ngrp  = threadIdx.x >> 5;       // 0..7 -> 4 column tiles each
    const int mtile = blockIdx.x;
    if (mtile * 16 >= nNodes) return;         // wave-uniform exit

    const int half = lane >> 4;
    const int l16  = lane & 15;

    int arow = mtile * 16 + l16;
    if (arow >= nNodes) arow = nNodes - 1;    // keep EXEC full for WMMA
    const __bf16* __restrict__ fb = featb + (size_t)arow * D_IN;

    // B columns for this wave's 4 tiles: ntile = ngrp*4 + t
    const int bcol0 = ngrp * 64 + l16;        // + t*16
    const __bf16* __restrict__ wb0 = Wb + (size_t)(bcol0)      * D_IN;
    const __bf16* __restrict__ wb1 = Wb + (size_t)(bcol0 + 16) * D_IN;
    const __bf16* __restrict__ wb2 = Wb + (size_t)(bcol0 + 32) * D_IN;
    const __bf16* __restrict__ wb3 = Wb + (size_t)(bcol0 + 48) * D_IN;

    v8f acc0 = {}, acc1 = {}, acc2 = {}, acc3 = {};
#pragma unroll
    for (int k0 = 0; k0 < D_IN; k0 += 32) {
        // A fragment: two contiguous 16B bf16 chunks per lane
        v8bf alo = *(const v8bf*)(fb + k0 + half * 16);
        v8bf ahi = *(const v8bf*)(fb + k0 + 8 + half * 16);
        v16bf A = __builtin_shufflevector(alo, ahi,
                   0, 1, 2, 3, 4, 5, 6, 7, 8, 9, 10, 11, 12, 13, 14, 15);
        // B fragments: 16 contiguous bf16 (32B, aligned) per tile
        v16bf B0 = *(const v16bf*)(wb0 + k0 + half * 16);
        v16bf B1 = *(const v16bf*)(wb1 + k0 + half * 16);
        v16bf B2 = *(const v16bf*)(wb2 + k0 + half * 16);
        v16bf B3 = *(const v16bf*)(wb3 + k0 + half * 16);

        acc0 = __builtin_amdgcn_wmma_f32_16x16x32_bf16(
                   false, A, false, B0, (short)0, acc0, false, false);
        acc1 = __builtin_amdgcn_wmma_f32_16x16x32_bf16(
                   false, A, false, B1, (short)0, acc1, false, false);
        acc2 = __builtin_amdgcn_wmma_f32_16x16x32_bf16(
                   false, A, false, B2, (short)0, acc2, false, false);
        acc3 = __builtin_amdgcn_wmma_f32_16x16x32_bf16(
                   false, A, false, B3, (short)0, acc3, false, false);
    }

#pragma unroll
    for (int t = 0; t < 4; ++t) {
        const v8f& acc = (t == 0) ? acc0 : (t == 1) ? acc1 : (t == 2) ? acc2 : acc3;
        const int bcol = bcol0 + t * 16;
        const float bias = (bcol < HD) ? bsrc[bcol] : bdst[bcol - HD];
#pragma unroll
        for (int r = 0; r < 8; ++r) {
            const int node = mtile * 16 + r + half * 8;
            if (node < nNodes)
                P[(size_t)node * NCOL + bcol] = acc[r] + bias;
        }
    }
}

// ---------------------------------------------------------------------------
// 4) Per-edge attention logits + segment max (ordered-int atomicMax).
//    One wave per edge; lane j-loop covers the 256 fused channels.
// ---------------------------------------------------------------------------
__global__ void __launch_bounds__(256)
score_kernel(const int* __restrict__ src, const int* __restrict__ dst,
             const float* __restrict__ P, const float* __restrict__ attn,
             float* __restrict__ score, int* __restrict__ maxbuf, int nEdges) {
    const int lane = threadIdx.x & 31;
    const int e    = blockIdx.x * 8 + (threadIdx.x >> 5);
    if (e >= nEdges) return;

    const size_t sb = (size_t)src[e] * NCOL;         // src projection base
    const size_t db = (size_t)dst[e] * NCOL + HD;    // dst projection base
    float sh[NHEAD] = {0.f, 0.f, 0.f, 0.f};
#pragma unroll
    for (int j = 0; j < 8; ++j) {
        const int c = j * 32 + lane;                 // channel = h*64 + d
        float v = P[sb + c] + P[db + c];
        v = (v >= 0.f) ? v : v * NEG_SLOPE;
        sh[j >> 1] += v * attn[c];
    }
#pragma unroll
    for (int h = 0; h < NHEAD; ++h)
#pragma unroll
        for (int off = 16; off > 0; off >>= 1)
            sh[h] += __shfl_xor(sh[h], off, 32);

    if (lane < NHEAD) {
        const float s = sh[lane];
        score[(size_t)e * NHEAD + lane] = s;
        int ord = __float_as_int(s);
        ord = (ord >= 0) ? ord : (ord ^ 0x7fffffff); // total-order encoding
        atomicMax(maxbuf + (size_t)dst[e] * NHEAD + lane, ord);
    }
}

// ---------------------------------------------------------------------------
// 5) expv = exp(score - max[dst]); accumulate denom[dst].
// ---------------------------------------------------------------------------
__global__ void exp_kernel(const int* __restrict__ dst,
                           const int* __restrict__ maxbuf,
                           float* __restrict__ score,
                           float* __restrict__ denom, int nEdges) {
    const int i = blockIdx.x * blockDim.x + threadIdx.x;
    if (i >= nEdges * NHEAD) return;
    const int e = i >> 2, h = i & 3;
    const int t = dst[e];
    int mo = maxbuf[(size_t)t * NHEAD + h];
    const float m = (mo >= 0) ? __int_as_float(mo)
                              : __int_as_float(mo ^ 0x7fffffff);
    const float v = expf(score[i] - m);
    score[i] = v;
    atomicAdd(denom + (size_t)t * NHEAD + h, v);
}

// ---------------------------------------------------------------------------
// 6) Weighted scatter: out[dst, c] += (expv/denom) * feat_src[src, c].
//    256 consecutive threads share one edge -> coalesced 1KB L2-hit reads.
// ---------------------------------------------------------------------------
__global__ void __launch_bounds__(256)
scatter_kernel(const int* __restrict__ src, const int* __restrict__ dst,
               const float* __restrict__ P, const float* __restrict__ score,
               const float* __restrict__ denom, float* __restrict__ out,
               int nEdges) {
    const size_t i = (size_t)blockIdx.x * blockDim.x + threadIdx.x;
    if (i >= (size_t)nEdges * HD) return;
    const int e = (int)(i >> 8);
    const int c = (int)(i & 255);
    const int h = c >> 6;
    const int t = dst[e];
    const float a = score[(size_t)e * NHEAD + h] /
                    denom[(size_t)t * NHEAD + h];
    atomicAdd(out + (size_t)t * HD + c, a * P[(size_t)src[e] * NCOL + c]);
}

// ---------------------------------------------------------------------------
extern "C" void kernel_launch(void* const* d_in, const int* in_sizes, int n_in,
                              void* d_out, int out_size, void* d_ws, size_t ws_size,
                              hipStream_t stream) {
    const float* feat = (const float*)d_in[0];
    const int*   src  = (const int*)  d_in[1];
    const int*   dst  = (const int*)  d_in[2];
    const float* Wsrc = (const float*)d_in[3];
    const float* bsrc = (const float*)d_in[4];
    const float* Wdst = (const float*)d_in[5];
    const float* bdst = (const float*)d_in[6];
    const float* attn = (const float*)d_in[7];
    float* out = (float*)d_out;

    const int nNodes = in_sizes[0] / D_IN;
    const int nEdges = in_sizes[1];

    // --- workspace carve-out (256B aligned) ---
    uint8_t* ws = (uint8_t*)d_ws;
    size_t off = 0;
    float* P = (float*)(ws + off);
    off += (size_t)nNodes * NCOL * sizeof(float);         off = (off + 255) & ~(size_t)255;
    __bf16* Wb = (__bf16*)(ws + off);
    off += (size_t)NCOL * D_IN * sizeof(__bf16);          off = (off + 255) & ~(size_t)255;
    __bf16* featb = (__bf16*)(ws + off);
    off += (size_t)nNodes * D_IN * sizeof(__bf16);        off = (off + 255) & ~(size_t)255;
    float* score = (float*)(ws + off);
    off += (size_t)nEdges * NHEAD * sizeof(float);        off = (off + 255) & ~(size_t)255;
    int* maxbuf = (int*)(ws + off);
    off += (size_t)nNodes * NHEAD * sizeof(int);          off = (off + 255) & ~(size_t)255;
    float* denom = (float*)(ws + off);

    // 1) weights + features -> bf16
    wconv_kernel<<<(NCOL * D_IN + 255) / 256, 256, 0, stream>>>(Wsrc, Wdst, Wb);
    const int nQuads = nNodes * D_IN / 4;
    fconv_kernel<<<(nQuads + 255) / 256, 256, 0, stream>>>(feat, featb, nQuads);
    // 2) init accumulators
    init_kernel<<<(nNodes * HD + 255) / 256, 256, 0, stream>>>(maxbuf, denom, out, nNodes);
    // 3) WMMA projection GEMM: one block per 16-row strip
    const int mtiles = (nNodes + 15) / 16;
    proj_wmma_kernel<<<mtiles, 256, 0, stream>>>(featb, Wb, bsrc, bdst, P, nNodes);
    // 4) attention logits + segment max
    score_kernel<<<(nEdges + 7) / 8, 256, 0, stream>>>(src, dst, P, attn, score, maxbuf, nEdges);
    // 5) exp + segment sum
    exp_kernel<<<(nEdges * NHEAD + 255) / 256, 256, 0, stream>>>(dst, maxbuf, score, denom, nEdges);
    // 6) weighted scatter-sum
    const size_t scat = (size_t)nEdges * HD;
    scatter_kernel<<<(unsigned)((scat + 255) / 256), 256, 0, stream>>>(src, dst, P, score, denom, out, nEdges);
}
